// GCL_basic_86620900426031
// MI455X (gfx1250) — matrix operand order repaired and checked by reading, not compiled, not run on hardware
//
#include <hip/hip_runtime.h>
#include <hip/hip_fp16.h>

typedef __attribute__((ext_vector_type(16))) _Float16 v16h;
typedef __attribute__((ext_vector_type(8)))  _Float16 v8h;
typedef __attribute__((ext_vector_type(8)))  float    v8f;

#define N_NODES 50000
#define N_EDGES 800000
#define DF 64
#define HID 64

union V16 { v16h v; v8h h[2]; };

__device__ __forceinline__ v8f wmma_f16(v16h a, v16h b, v8f c) {
  // D = A(16x32) * B(32x16) + C, f32 accumulate
  return __builtin_amdgcn_wmma_f32_16x16x32_f16(
      /*neg_a=*/false, a, /*neg_b=*/false, b,
      /*c_mod=*/(short)0, c, /*reuse_a=*/false, /*reuse_b=*/false);
}

__device__ __forceinline__ float silu_f(float x) {
  return x * (1.0f / (1.0f + __expf(-x)));
}

// Pre-swizzle W^T (W is row-major [K][64]) into WMMA A-fragment order in LDS.
// Fragment f = mc*kchunks + kc covers A tile [16 out-dims x 32 in-dims].
// Layout: frag[f][lane][j] (16 f16 = 32B per lane).
__device__ void build_frags(const float* __restrict__ W, int kchunks, int fbase,
                            unsigned short* lds) {
  const int total = 4 * kchunks * 512;
  for (int idx = threadIdx.x; idx < total; idx += blockDim.x) {
    const int f    = idx >> 9;
    const int r    = idx & 511;
    const int lane = r >> 4;
    const int j    = r & 15;
    const int mc = f / kchunks;
    const int kc = f - mc * kchunks;
    const int m = mc * 16 + (lane & 15);
    const int kbase = kc * 32 + ((lane & 16) ? 16 : 0);
    const int k = kbase + j;   // K=0..15 (lo half) / 16..31 (hi half)
    _Float16 hv = (_Float16)W[k * 64 + m];           // W^T[m][k] == W[k][m]
    lds[(size_t)(fbase + f) * 512 + r] = __builtin_bit_cast(unsigned short, hv);
  }
}

__device__ __forceinline__ v16h load_frag(const unsigned short* lds, int f, int lane) {
  V16 t;
  const v8h* p = (const v8h*)(lds + (size_t)f * 512 + lane * 16);
  t.h[0] = p[0];
  t.h[1] = p[1];
  return t.v;
}

// 16 contiguous f32 -> v16h (B-operand fragment: per-lane contiguous K-run)
__device__ __forceinline__ v16h load_feat16(const float* __restrict__ p) {
  const float4* q = (const float4*)p;
  float4 a = q[0], b = q[1], c = q[2], d = q[3];
  v16h r;
  r[0]=(_Float16)a.x;  r[1]=(_Float16)a.y;  r[2]=(_Float16)a.z;  r[3]=(_Float16)a.w;
  r[4]=(_Float16)b.x;  r[5]=(_Float16)b.y;  r[6]=(_Float16)b.z;  r[7]=(_Float16)b.w;
  r[8]=(_Float16)c.x;  r[9]=(_Float16)c.y;  r[10]=(_Float16)c.z; r[11]=(_Float16)c.w;
  r[12]=(_Float16)d.x; r[13]=(_Float16)d.y; r[14]=(_Float16)d.z; r[15]=(_Float16)d.w;
  return r;
}

__device__ __forceinline__ void load8(const float* __restrict__ p, float out[8]) {
  float4 a = ((const float4*)p)[0];
  float4 b = ((const float4*)p)[1];
  out[0]=a.x; out[1]=a.y; out[2]=a.z; out[3]=a.w;
  out[4]=b.x; out[5]=b.y; out[6]=b.z; out[7]=b.w;
}

// Build layer-2 B fragment (h^T tile, K in registers, edge/node in lane)
// from the layer-1 accumulator layout via half-wave exchange only.
__device__ __forceinline__ v16h build_bfrag(const float hval[4][8], int kc, bool hi) {
  V16 t;
#pragma unroll
  for (int j = 0; j < 8; ++j) {
    float lo_v = hval[2 * kc + 0][j];
    float hi_v = hval[2 * kc + 1][j];
    float sw_lo = __shfl_xor(lo_v, 16, 32);
    float sw_hi = __shfl_xor(hi_v, 16, 32);
    t.v[j]     = (_Float16)(hi ? sw_hi : lo_v);
    t.v[j + 8] = (_Float16)(hi ? hi_v  : sw_lo);
  }
  return t.v;
}

// ---------------------------------------------------------------------------
// Edge kernel: edge_feat = silu(silu([x[row],x[col],attr]@We1+b1)@We2+b2)*mask
// plus atomic scatter into agg[row]. One wave handles 16 edges.
// ---------------------------------------------------------------------------
__global__ void __launch_bounds__(256, 2) gnn_edge_kernel(
    const float* __restrict__ x, const int* __restrict__ eidx,
    const float* __restrict__ emask, const float* __restrict__ eattr,
    const float* __restrict__ We1, const float* __restrict__ be1,
    const float* __restrict__ We2, const float* __restrict__ be2,
    float* __restrict__ edge_feat, float* __restrict__ agg) {
  __shared__ __align__(16) unsigned short fragLDS[24 * 512];
  build_frags(We1, 4, 0, fragLDS);    // 16 frags: We1^T (K = 128 feature dims)
  build_frags(We2, 2, 16, fragLDS);   //  8 frags: We2^T (K = 64 hidden dims)
  __syncthreads();

  const int lane = threadIdx.x & 31;
  const int wave = threadIdx.x >> 5;
  const int tile = blockIdx.x * 8 + wave;
  const int et = tile * 16;
  if (et >= N_EDGES) return;

  const bool hi = (lane & 16) != 0;
  const int koff = hi ? 16 : 0;   // K offset inside 32-chunk for B operand
  const int hoff = hi ? 8 : 0;    // M offset inside 16-chunk for C/D layout
  const int e = et + (lane & 15);
  const int rI = eidx[e];
  const int cI = eidx[N_EDGES + e];
  const float attrv = eattr[e];
  const float maskv = emask[e];

  // ----- layer 1: h^T = We1^T(64x128) @ e_in^T(128x16) -----
  v8f hacc[4] = {};
#pragma unroll
  for (int kc = 0; kc < 4; ++kc) {
    const float* p = (kc < 2)
        ? (x + (size_t)rI * DF + kc * 32 + koff)
        : (x + (size_t)cI * DF + (kc - 2) * 32 + koff);
    v16h b = load_feat16(p);
#pragma unroll
    for (int mc = 0; mc < 4; ++mc) {
      v16h a = load_frag(fragLDS, mc * 4 + kc, lane);
      hacc[mc] = wmma_f16(a, b, hacc[mc]);
    }
  }

  // epilogue 1: + be1 + attr * We1[row 128], SiLU
  float hval[4][8];
#pragma unroll
  for (int mc = 0; mc < 4; ++mc) {
    const int base = mc * 16 + hoff;
    float bb[8], ww[8];
    load8(be1 + base, bb);
    load8(We1 + 128 * 64 + base, ww);
#pragma unroll
    for (int g = 0; g < 8; ++g) {
      float pre = hacc[mc][g] + bb[g] + attrv * ww[g];
      hval[mc][g] = silu_f(pre);
    }
  }

  // ----- layer 2: out^T = We2^T(64x64) @ h^T(64x16) -----
  v16h bf0 = build_bfrag(hval, 0, hi);
  v16h bf1 = build_bfrag(hval, 1, hi);
  v8f oacc[4] = {};
#pragma unroll
  for (int mc = 0; mc < 4; ++mc) {
    v16h a0 = load_frag(fragLDS, 16 + mc * 2 + 0, lane);
    oacc[mc] = wmma_f16(a0, bf0, oacc[mc]);
    v16h a1 = load_frag(fragLDS, 16 + mc * 2 + 1, lane);
    oacc[mc] = wmma_f16(a1, bf1, oacc[mc]);
  }

  // epilogue 2: + be2, SiLU, * mask; store edge_feat row; scatter into agg
#pragma unroll
  for (int mc = 0; mc < 4; ++mc) {
    const int base = mc * 16 + hoff;
    float bb[8];
    load8(be2 + base, bb);
    float outv[8];
#pragma unroll
    for (int g = 0; g < 8; ++g)
      outv[g] = silu_f(oacc[mc][g] + bb[g]) * maskv;

    float* dst = edge_feat + (size_t)e * HID + base;
    ((float4*)dst)[0] = make_float4(outv[0], outv[1], outv[2], outv[3]);
    ((float4*)dst)[1] = make_float4(outv[4], outv[5], outv[6], outv[7]);

    float* ag = agg + (size_t)rI * HID + base;
#pragma unroll
    for (int g = 0; g < 8; ++g)
      __hip_atomic_fetch_add(ag + g, outv[g], __ATOMIC_RELAXED,
                             __HIP_MEMORY_SCOPE_AGENT);
  }
}

// ---------------------------------------------------------------------------
// Node kernel: x_out = x + silu([x,agg]@Wn1+bn1)@Wn2 + bn2. One wave = 16 nodes.
// ---------------------------------------------------------------------------
__global__ void __launch_bounds__(256, 2) gnn_node_kernel(
    const float* __restrict__ x, const float* __restrict__ agg,
    const float* __restrict__ Wn1, const float* __restrict__ bn1,
    const float* __restrict__ Wn2, const float* __restrict__ bn2,
    float* __restrict__ x_out) {
  __shared__ __align__(16) unsigned short fragLDS[24 * 512];
  build_frags(Wn1, 4, 0, fragLDS);
  build_frags(Wn2, 2, 16, fragLDS);
  __syncthreads();

  const int lane = threadIdx.x & 31;
  const int wave = threadIdx.x >> 5;
  const int tile = blockIdx.x * 8 + wave;
  const int nt = tile * 16;
  if (nt >= N_NODES) return;

  const bool hi = (lane & 16) != 0;
  const int koff = hi ? 16 : 0;
  const int hoff = hi ? 8 : 0;
  const int n = nt + (lane & 15);

  v8f hacc[4] = {};
#pragma unroll
  for (int kc = 0; kc < 4; ++kc) {
    const float* p = (kc < 2)
        ? (x + (size_t)n * DF + kc * 32 + koff)
        : (agg + (size_t)n * HID + (kc - 2) * 32 + koff);
    v16h b = load_feat16(p);
#pragma unroll
    for (int mc = 0; mc < 4; ++mc) {
      v16h a = load_frag(fragLDS, mc * 4 + kc, lane);
      hacc[mc] = wmma_f16(a, b, hacc[mc]);
    }
  }

  float hval[4][8];
#pragma unroll
  for (int mc = 0; mc < 4; ++mc) {
    const int base = mc * 16 + hoff;
    float bb[8];
    load8(bn1 + base, bb);
#pragma unroll
    for (int g = 0; g < 8; ++g)
      hval[mc][g] = silu_f(hacc[mc][g] + bb[g]);
  }

  v16h bf0 = build_bfrag(hval, 0, hi);
  v16h bf1 = build_bfrag(hval, 1, hi);
  v8f oacc[4] = {};
#pragma unroll
  for (int mc = 0; mc < 4; ++mc) {
    v16h a0 = load_frag(fragLDS, 16 + mc * 2 + 0, lane);
    oacc[mc] = wmma_f16(a0, bf0, oacc[mc]);
    v16h a1 = load_frag(fragLDS, 16 + mc * 2 + 1, lane);
    oacc[mc] = wmma_f16(a1, bf1, oacc[mc]);
  }

#pragma unroll
  for (int mc = 0; mc < 4; ++mc) {
    const int base = mc * 16 + hoff;
    float bb[8], xr[8];
    load8(bn2 + base, bb);
    load8(x + (size_t)n * DF + base, xr);
    float outv[8];
#pragma unroll
    for (int g = 0; g < 8; ++g)
      outv[g] = xr[g] + oacc[mc][g] + bb[g];
    float* dst = x_out + (size_t)n * DF + base;
    ((float4*)dst)[0] = make_float4(outv[0], outv[1], outv[2], outv[3]);
    ((float4*)dst)[1] = make_float4(outv[4], outv[5], outv[6], outv[7]);
  }
}

__global__ void zero_kernel(float4* __restrict__ p, int n4) {
  int i = blockIdx.x * blockDim.x + threadIdx.x;
  if (i < n4) p[i] = make_float4(0.f, 0.f, 0.f, 0.f);
}

extern "C" void kernel_launch(void* const* d_in, const int* in_sizes, int n_in,
                              void* d_out, int out_size, void* d_ws, size_t ws_size,
                              hipStream_t stream) {
  const float* x     = (const float*)d_in[0];
  const int*   eidx  = (const int*)d_in[1];
  const float* emask = (const float*)d_in[2];
  const float* eattr = (const float*)d_in[3];
  const float* We1   = (const float*)d_in[4];
  const float* be1   = (const float*)d_in[5];
  const float* We2   = (const float*)d_in[6];
  const float* be2   = (const float*)d_in[7];
  const float* Wn1   = (const float*)d_in[8];
  const float* bn1   = (const float*)d_in[9];
  const float* Wn2   = (const float*)d_in[10];
  const float* bn2   = (const float*)d_in[11];

  float* x_out     = (float*)d_out;                          // [N, 64]
  float* edge_feat = (float*)d_out + (size_t)N_NODES * DF;   // [E, 64]
  float* agg       = (float*)d_ws;                           // [N, 64] scratch

  // zero the scatter accumulator
  const int n4 = (N_NODES * HID) / 4;
  zero_kernel<<<(n4 + 255) / 256, 256, 0, stream>>>((float4*)agg, n4);

  // edge MLP + scatter: 8 waves/block * 16 edges/wave = 128 edges/block
  const int edge_blocks = N_EDGES / 128;  // 800000 / 128 = 6250 exact
  gnn_edge_kernel<<<edge_blocks, 256, 0, stream>>>(
      x, eidx, emask, eattr, We1, be1, We2, be2, edge_feat, agg);

  // node MLP + residual
  const int node_tiles = (N_NODES + 15) / 16;
  const int node_blocks = (node_tiles + 7) / 8;
  gnn_node_kernel<<<node_blocks, 256, 0, stream>>>(
      x, agg, Wn1, bn1, Wn2, bn2, x_out);
}
